// Qwen3SelectiveModel_5214090297942
// MI455X (gfx1250) — matrix operand order repaired and checked by reading, not compiled, not run on hardware
//
#include <hip/hip_runtime.h>
#include <hip/hip_bf16.h>
#include <math.h>

// ---- model config (matches reference) ----
#define LAYERS 2
#define RQ 4
#define SEQ 1024
#define HID 1024
#define NHq 16
#define NKVq 8
#define DHq 128
#define FFq 3072
#define TOKENS (RQ * SEQ)     // 4096
#define QDIM (NHq * DHq)      // 2048
#define KVDIM (NKVq * DHq)    // 1024
#define EPSF 1e-6f
#define ROPE_THETA 1.0e6f

typedef _Float16 v16h __attribute__((ext_vector_type(16)));
typedef _Float16 v8h  __attribute__((ext_vector_type(8)));
typedef float    v8f  __attribute__((ext_vector_type(8)));
typedef int      ivec4 __attribute__((vector_size(16)));  // matches builtin param type

#if __has_builtin(__builtin_amdgcn_global_load_async_to_lds_b128)
#define USE_ASYNC_LDS 1
#else
#define USE_ASYNC_LDS 0
#endif

#if USE_ASYNC_LDS
__device__ __forceinline__ void wait_async0() {
#if __has_builtin(__builtin_amdgcn_s_wait_asynccnt)
  __builtin_amdgcn_s_wait_asynccnt(0);
#else
  asm volatile("s_wait_asynccnt 0" ::: "memory");
#endif
}
#endif

__device__ __forceinline__ v8f wmma16(v16h a, v16h b, v8f c) {
  return __builtin_amdgcn_wmma_f32_16x16x32_f16(false, a, false, b, (short)0, c,
                                                false, false);
}

__device__ __forceinline__ v16h combine16(v8h lo, v8h hi) {
  v16h a;
#pragma unroll
  for (int i = 0; i < 8; ++i) { a[i] = lo[i]; a[i + 8] = hi[i]; }
  return a;
}

// A fragment (16x32 f16) from f16 row-major [*, lda]; lda mult of 8, kc of 32.
__device__ __forceinline__ v16h load_a_frag16(const _Float16* __restrict__ A, int lda,
                                              int row0, int kc, int lane) {
  const int r  = lane & 15;
  const int ko = (lane >> 4) << 4;
  const _Float16* p = A + (size_t)(row0 + r) * lda + kc;
  return combine16(*(const v8h*)(p + ko), *(const v8h*)(p + ko + 8));
}

// B fragment where the lane's 16 consecutive K-values are contiguous at p (32B aligned)
__device__ __forceinline__ v16h load_b_contig16(const _Float16* __restrict__ p) {
  return combine16(*(const v8h*)p, *(const v8h*)(p + 8));
}

// ---------------- f32 -> f16 convert + transpose: src[K][N] -> dst[N][K] ----------------
__global__ void f32_to_f16_t_kernel(const float* __restrict__ src,
                                    _Float16* __restrict__ dst, int K, int N) {
  const int i = blockIdx.x * 256 + threadIdx.x;
  if (i < K * N) {
    const int k = i / N, n = i % N;
    dst[(size_t)n * K + k] = (_Float16)src[i];
  }
}

// ---------------- GEMM: C[M,N] = A[M,K](f16) x BT[N][K](f16, pre-transposed) ------------
// 128 threads = 4 waves; block tile 128x64; each wave owns a 32x64 C tile.
// B tile (64 n-rows x 32 k) double-buffered in LDS via async-to-LDS copies.
#define BK 32
#define BSTR 40  // padded LDS row stride (halfs)
__global__ void gemm_wmma_kernel(const _Float16* __restrict__ A,
                                 const _Float16* __restrict__ BT,
                                 const float* __restrict__ addend,
                                 float* __restrict__ Cf, _Float16* __restrict__ Ch,
                                 int M, int N, int K) {
  __shared__ __align__(16) _Float16 Bs[2][64][BSTR];
  const int tid  = threadIdx.x;
  const int lane = tid & 31;
  const int wave = tid >> 5;
  const int m0   = blockIdx.x * 128 + wave * 32;
  const int nblk = blockIdx.y * 64;
  const int ln   = tid >> 1;          // n row 0..63
  const int lk   = (tid & 1) * 16;    // k chunk 0 or 16
  const int n16  = lane & 15;
  const int kb   = (lane >> 4) << 4;
  v8f acc[2][4] = {};

  auto issue_tile = [&](int kc, int buf) {
    const _Float16* src = BT + (size_t)(nblk + ln) * K + kc + lk;
#if USE_ASYNC_LDS
    __builtin_amdgcn_global_load_async_to_lds_b128(
        (ivec4*)src, (ivec4*)&Bs[buf][ln][lk], 0, 0);
    __builtin_amdgcn_global_load_async_to_lds_b128(
        (ivec4*)(src + 8), (ivec4*)&Bs[buf][ln][lk + 8], 0, 0);
#else
    v8h w0 = *(const v8h*)src;
    v8h w1 = *(const v8h*)(src + 8);
    *(v8h*)&Bs[buf][ln][lk]     = w0;
    *(v8h*)&Bs[buf][ln][lk + 8] = w1;
    if (kc + BK < K) __builtin_prefetch(src + BK, 0, 3);
#endif
  };

  issue_tile(0, 0);
  int buf = 0;
  for (int kc = 0; kc < K; kc += BK) {
#if USE_ASYNC_LDS
    wait_async0();
#endif
    __syncthreads();
    if (kc + BK < K) issue_tile(kc + BK, buf ^ 1);
    v16h a0 = load_a_frag16(A, K, m0,      kc, lane);
    v16h a1 = load_a_frag16(A, K, m0 + 16, kc, lane);
#pragma unroll
    for (int t = 0; t < 4; ++t) {
      v16h b = combine16(*(const v8h*)&Bs[buf][t * 16 + n16][kb],
                         *(const v8h*)&Bs[buf][t * 16 + n16][kb + 8]);
      acc[0][t] = wmma16(a0, b, acc[0][t]);
      acc[1][t] = wmma16(a1, b, acc[1][t]);
    }
    buf ^= 1;
#if !USE_ASYNC_LDS
    __syncthreads();
#endif
  }

  const int hg = lane >> 4, col = lane & 15;
#pragma unroll
  for (int w = 0; w < 2; ++w)
#pragma unroll
    for (int t = 0; t < 4; ++t)
#pragma unroll
      for (int s = 0; s < 8; ++s) {
        const int row = m0 + w * 16 + s + 8 * hg;
        const size_t idx = (size_t)row * N + nblk + 16 * t + col;
        float val = acc[w][t][s];
        if (Cf) {
          if (addend) val += addend[idx];
          Cf[idx] = val;
        } else {
          Ch[idx] = (_Float16)val;
        }
      }
}

// ---------------- row RMSNorm (fp32 in; fp32 or f16 out) ----------------
__global__ void rmsnorm_kernel(const float* __restrict__ x, const float* __restrict__ w,
                               float* __restrict__ outf, _Float16* __restrict__ outh,
                               int N) {
  __shared__ float red[256];
  const int row = blockIdx.x, tid = threadIdx.x;
  const float* xr = x + (size_t)row * N;
  float s = 0.f;
  for (int i = tid; i < N; i += 256) { float v = xr[i]; s += v * v; }
  red[tid] = s;
  __syncthreads();
  for (int off = 128; off > 0; off >>= 1) {
    if (tid < off) red[tid] += red[tid + off];
    __syncthreads();
  }
  const float inv = rsqrtf(red[0] / (float)N + EPSF);
  for (int i = tid; i < N; i += 256) {
    const float v = xr[i] * inv * w[i];
    if (outf) outf[(size_t)row * N + i] = v;
    else      outh[(size_t)row * N + i] = (_Float16)v;
  }
}

// ------- per-head RMSNorm (DH) + RoPE, in place on f16; scale folded in -------
__global__ void qknorm_rope_kernel(_Float16* __restrict__ x, const float* __restrict__ w,
                                   int heads, float scale) {
  __shared__ float red[DHq];
  __shared__ float vn[DHq];
  const int tok = blockIdx.x, head = blockIdx.y, d = threadIdx.x;
  _Float16* base = x + (size_t)tok * heads * DHq + (size_t)head * DHq;
  const float v = (float)base[d];
  red[d] = v * v;
  __syncthreads();
  for (int off = 64; off > 0; off >>= 1) {
    if (d < off) red[d] += red[d + off];
    __syncthreads();
  }
  const float inv = rsqrtf(red[0] / (float)DHq + EPSF);
  vn[d] = v * inv * w[d];
  __syncthreads();
  const int dp = (d < 64) ? d + 64 : d - 64;
  const float rot = (d < 64) ? -vn[dp] : vn[dp];
  const float pos = (float)(tok % SEQ);
  const float invf = __powf(ROPE_THETA, -((float)((d & 63) * 2)) / (float)DHq);
  const float ang = pos * invf;
  base[d] = (_Float16)((vn[d] * __cosf(ang) + rot * __sinf(ang)) * scale);
}

// ------- V transpose: [tok][kv*DH+d] -> Vt [r][kv][d][s] (all f16) -------
__global__ void transpose_v_kernel(const _Float16* __restrict__ v,
                                   _Float16* __restrict__ vt) {
  const int s  = blockIdx.x * blockDim.x + threadIdx.x;
  const int d  = blockIdx.y;
  const int rk = blockIdx.z;
  const int r = rk / NKVq, kv = rk % NKVq;
  vt[((size_t)rk * DHq + d) * SEQ + s] =
      v[(size_t)(r * SEQ + s) * KVDIM + kv * DHq + d];
}

// ---------------- SiLU(g) * u (f16 in/out, fp32 math) ----------------
__global__ void silu_mul_kernel(_Float16* __restrict__ g, const _Float16* __restrict__ u,
                                int n) {
  const int i = blockIdx.x * blockDim.x + threadIdx.x;
  if (i < n) {
    const float gv = (float)g[i];
    g[i] = (_Float16)((gv / (1.f + __expf(-gv))) * (float)u[i]);
  }
}

// ---------------- fused causal attention (flash-style, GQA, f16 WMMA) ----------------
__global__ void attention_kernel(const _Float16* __restrict__ q,
                                 const _Float16* __restrict__ kmat,
                                 const _Float16* __restrict__ vt,
                                 _Float16* __restrict__ out) {
  __shared__ float ldsP[16][32];
  const int lane = threadIdx.x;
  const int m0   = blockIdx.x * 16;
  const int h    = blockIdx.y;
  const int r    = blockIdx.z;
  const int kvh  = h / (NHq / NKVq);
  const int tok0 = r * SEQ + m0;
  const int hg = lane >> 4, col = lane & 15;

  v16h qf[4];
#pragma unroll
  for (int c = 0; c < 4; ++c)
    qf[c] = load_a_frag16(q, QDIM, tok0, h * DHq + c * 32, lane);

  v8f o[8] = {};
  float row_max[8], row_sum[8];
#pragma unroll
  for (int s = 0; s < 8; ++s) { row_max[s] = -1e30f; row_sum[s] = 0.f; }

  for (int j0 = 0; j0 <= m0; j0 += 32) {
    v8f s0 = {}, s1 = {};
#pragma unroll
    for (int c = 0; c < 4; ++c) {
      const _Float16* p0 = kmat + (size_t)(r * SEQ + j0 + col) * KVDIM
                                + kvh * DHq + c * 32 + (hg << 4);
      v16h b0 = load_b_contig16(p0);
      v16h b1 = load_b_contig16(p0 + (size_t)16 * KVDIM);
      s0 = wmma16(qf[c], b0, s0);
      s1 = wmma16(qf[c], b1, s1);
    }

    float pm0[8], pm1[8], alpha[8];
#pragma unroll
    for (int s = 0; s < 8; ++s) {
      const int qidx = m0 + s + 8 * hg;
      float x0 = (j0 + col      > qidx) ? -1e30f : s0[s];
      float x1 = (j0 + 16 + col > qidx) ? -1e30f : s1[s];
      float mx = fmaxf(x0, x1);
#pragma unroll
      for (int off = 1; off < 16; off <<= 1)
        mx = fmaxf(mx, __shfl_xor(mx, off, 16));
      const float mn = fmaxf(row_max[s], mx);
      const float a  = __expf(row_max[s] - mn);
      const float p0 = __expf(x0 - mn);
      const float p1 = __expf(x1 - mn);
      float sm = p0 + p1;
#pragma unroll
      for (int off = 1; off < 16; off <<= 1)
        sm += __shfl_xor(sm, off, 16);
      row_sum[s] = row_sum[s] * a + sm;
      row_max[s] = mn;
      alpha[s] = a; pm0[s] = p0; pm1[s] = p1;
    }
#pragma unroll
    for (int t = 0; t < 8; ++t)
#pragma unroll
      for (int s = 0; s < 8; ++s) o[t][s] *= alpha[s];

    __syncthreads();
#pragma unroll
    for (int s = 0; s < 8; ++s) {
      ldsP[s + 8 * hg][col]      = pm0[s];
      ldsP[s + 8 * hg][16 + col] = pm1[s];
    }
    __syncthreads();
    v16h pa;
    {
      const int ko = hg << 4;
#pragma unroll
      for (int i = 0; i < 8; ++i) pa[i]     = (_Float16)ldsP[col][ko + i];
#pragma unroll
      for (int i = 0; i < 8; ++i) pa[i + 8] = (_Float16)ldsP[col][ko + 8 + i];
    }

#pragma unroll
    for (int t = 0; t < 8; ++t) {
      const _Float16* pv = vt + ((size_t)(r * NKVq + kvh) * DHq + t * 16 + col) * SEQ
                              + j0 + (hg << 4);
      o[t] = wmma16(pa, load_b_contig16(pv), o[t]);
    }
  }

#pragma unroll
  for (int t = 0; t < 8; ++t)
#pragma unroll
    for (int s = 0; s < 8; ++s) {
      const int row = tok0 + s + 8 * hg;
      out[(size_t)row * QDIM + h * DHq + t * 16 + col] =
          (_Float16)(o[t][s] / row_sum[s]);
    }
}

// ---------------- orchestration ----------------
extern "C" void kernel_launch(void* const* d_in, const int* in_sizes, int n_in,
                              void* d_out, int out_size, void* d_ws, size_t ws_size,
                              hipStream_t stream) {
  (void)in_sizes; (void)n_in; (void)out_size; (void)ws_size;
  const float* hs  = (const float*)d_in[0];
  const float* Wq  = (const float*)d_in[1];
  const float* Wk  = (const float*)d_in[2];
  const float* Wv  = (const float*)d_in[3];
  const float* Wo  = (const float*)d_in[4];
  const float* qnw = (const float*)d_in[5];
  const float* knw = (const float*)d_in[6];
  const float* ln1 = (const float*)d_in[7];
  const float* ln2 = (const float*)d_in[8];
  const float* Wg  = (const float*)d_in[9];
  const float* Wu  = (const float*)d_in[10];
  const float* Wd  = (const float*)d_in[11];
  const float* fnw = (const float*)d_in[12];
  float* out = (float*)d_out;

  float* x = (float*)d_ws;                                   // [4096,1024] f32
  _Float16* hws = (_Float16*)(x + (size_t)TOKENS * HID);
  const size_t NWQ = (size_t)HID * QDIM, NWK = (size_t)HID * KVDIM,
               NWV = (size_t)HID * KVDIM, NWO = (size_t)QDIM * HID,
               NWG = (size_t)HID * FFq,  NWU = (size_t)HID * FFq,
               NWD = (size_t)FFq * HID;
  _Float16* wq16 = hws;             // transposed: [QDIM][HID]
  _Float16* wk16 = wq16 + NWQ;      // [KVDIM][HID]
  _Float16* wv16 = wk16 + NWK;      // [KVDIM][HID]
  _Float16* wo16 = wv16 + NWV;      // [HID][QDIM]
  _Float16* wg16 = wo16 + NWO;      // [FF][HID]
  _Float16* wu16 = wg16 + NWG;      // [FF][HID]
  _Float16* wd16 = wu16 + NWU;      // [HID][FF]
  _Float16* hbuf = wd16 + NWD;                     // [4096,1024]
  _Float16* qh   = hbuf + (size_t)TOKENS * HID;    // [4096,2048]
  _Float16* kh   = qh   + (size_t)TOKENS * QDIM;   // [4096,1024]
  _Float16* vh   = kh   + (size_t)TOKENS * KVDIM;  // [4096,1024]
  _Float16* vtb  = vh   + (size_t)TOKENS * KVDIM;  // [R,NKV,DH,S]
  _Float16* aoh  = vtb  + (size_t)TOKENS * KVDIM;  // [4096,2048]
  _Float16* gh = qh;   // MLP reuse (regions dead by then)
  _Float16* uh = vh;

  (void)hipMemcpyAsync(x, hs, sizeof(float) * (size_t)TOKENS * HID,
                       hipMemcpyDeviceToDevice, stream);

  const dim3 b32(32), b128(DHq), b256(256), bg(128);
  const float qscale = 0.08838834764831845f;  // 1/sqrt(128)

#define CONVT(srcp, dstp, Kd, Nd)                                                \
  f32_to_f16_t_kernel<<<(int)((((size_t)(Kd) * (Nd)) + 255) / 256), b256, 0,     \
                        stream>>>(srcp, dstp, (Kd), (Nd))

  for (int l = 0; l < LAYERS; ++l) {
    CONVT(Wq + l * NWQ, wq16, HID,  QDIM);
    CONVT(Wk + l * NWK, wk16, HID,  KVDIM);
    CONVT(Wv + l * NWV, wv16, HID,  KVDIM);
    CONVT(Wo + l * NWO, wo16, QDIM, HID);
    CONVT(Wg + l * NWG, wg16, HID,  FFq);
    CONVT(Wu + l * NWU, wu16, HID,  FFq);
    CONVT(Wd + l * NWD, wd16, FFq,  HID);

    rmsnorm_kernel<<<TOKENS, b256, 0, stream>>>(x, ln1 + (size_t)l * HID, nullptr, hbuf, HID);
    gemm_wmma_kernel<<<dim3(TOKENS / 128, QDIM / 64), bg, 0, stream>>>(
        hbuf, wq16, nullptr, nullptr, qh, TOKENS, QDIM, HID);
    gemm_wmma_kernel<<<dim3(TOKENS / 128, KVDIM / 64), bg, 0, stream>>>(
        hbuf, wk16, nullptr, nullptr, kh, TOKENS, KVDIM, HID);
    gemm_wmma_kernel<<<dim3(TOKENS / 128, KVDIM / 64), bg, 0, stream>>>(
        hbuf, wv16, nullptr, nullptr, vh, TOKENS, KVDIM, HID);
    qknorm_rope_kernel<<<dim3(TOKENS, NHq),  b128, 0, stream>>>(qh, qnw + (size_t)l * DHq, NHq, qscale);
    qknorm_rope_kernel<<<dim3(TOKENS, NKVq), b128, 0, stream>>>(kh, knw + (size_t)l * DHq, NKVq, 1.0f);
    transpose_v_kernel<<<dim3(SEQ / 256, DHq, RQ * NKVq), b256, 0, stream>>>(vh, vtb);
    attention_kernel<<<dim3(SEQ / 16, NHq, RQ), b32, 0, stream>>>(qh, kh, vtb, aoh);
    gemm_wmma_kernel<<<dim3(TOKENS / 128, HID / 64), bg, 0, stream>>>(
        aoh, wo16, x, x, nullptr, TOKENS, HID, QDIM);
    rmsnorm_kernel<<<TOKENS, b256, 0, stream>>>(x, ln2 + (size_t)l * HID, nullptr, hbuf, HID);
    gemm_wmma_kernel<<<dim3(TOKENS / 128, FFq / 64), bg, 0, stream>>>(
        hbuf, wg16, nullptr, nullptr, gh, TOKENS, FFq, HID);
    gemm_wmma_kernel<<<dim3(TOKENS / 128, FFq / 64), bg, 0, stream>>>(
        hbuf, wu16, nullptr, nullptr, uh, TOKENS, FFq, HID);
    silu_mul_kernel<<<(TOKENS * FFq + 255) / 256, b256, 0, stream>>>(gh, uh, TOKENS * FFq);
    gemm_wmma_kernel<<<dim3(TOKENS / 128, HID / 64), bg, 0, stream>>>(
        gh, wd16, x, x, nullptr, TOKENS, HID, FFq);
  }
  rmsnorm_kernel<<<TOKENS, b256, 0, stream>>>(x, fnw, out, nullptr, HID);
#undef CONVT
}
